// RSWABlock_40836549050747
// MI455X (gfx1250) — matrix operand order repaired and checked by reading, not compiled, not run on hardware
//
#include <hip/hip_runtime.h>
#include <hip/hip_bf16.h>
#include <math.h>

#define CCH 192
#define PIX 50176          // 224*224
#define NWIN 3136          // 4 * 28 * 28
#define TOK 64

typedef __attribute__((ext_vector_type(16))) _Float16 v16h;
typedef __attribute__((ext_vector_type(8)))  float    v8f;

__device__ inline v8f zero8() {
  v8f z;
#pragma unroll
  for (int i = 0; i < 8; ++i) z[i] = 0.0f;
  return z;
}

__device__ inline v8f wmma_f16(v16h a, v16h b, v8f c) {
  return __builtin_amdgcn_wmma_f32_16x16x32_f16(false, a, false, b, (short)0, c,
                                                false, false);
}

// ---------- packed fragment layouts (16x16x32 f16 WMMA, wave32) ----------
// One 16x32 (A) or 32x16 (B) tile = 512 halfs: lane L owns halfs
// [tile*512 + L*16 .. +15] -> a single contiguous 32B load per lane.
//
// A-layout element (m,k) of tile: lane = ((k>>4)&1)<<4 | (m&15),
//                                 i = k&15
// B-layout element (k,n) of tile: lane = ((k>>4)&1)<<4 | (n&15), i = k&15
__device__ inline int posA(int Kt, int t, int c) {
  return ((((t >> 4) * Kt) + (c >> 5)) << 9) +
         (((((c >> 4) & 1) << 4) | (t & 15)) << 4) +
         (c & 15);
}
__device__ inline int posB(int Nt, int k, int n) {
  return ((((k >> 5) * Nt) + (n >> 4)) << 9) +
         (((((k >> 4) & 1) << 4) | (n & 15)) << 4) + (k & 15);
}
__device__ inline v16h ldfrag(const _Float16* tile, int lane) {
  return *(const v16h*)(tile + (lane << 4));
}

// B fragment straight from strided f32 global (used only for x in k_pre1)
__device__ inline v16h ldb_g32(const float* b, int ldb, int lane) {
  v16h f;
  int n = lane & 15;
  int kb = (lane >> 4) << 4;
#pragma unroll
  for (int i = 0; i < 16; ++i) f[i] = (_Float16)b[(kb + i) * ldb + n];
  return f;
}

// ============ K0a/K0b: one-time weight pack (f32 -> swizzled f16) ============
__global__ __launch_bounds__(256) void k_packB(const float* __restrict__ w,
                                               _Float16* __restrict__ d,
                                               int K, int N) {
  int Nt = N >> 4;
  int idx = blockIdx.x * 256 + threadIdx.x;
  if (idx >= K * N) return;
  int i = idx & 15;
  int lane = (idx >> 4) & 31;
  int tile = idx >> 9;
  int nt = tile % Nt, kt = tile / Nt;
  int k = kt * 32 + ((lane >> 4) << 4) + i;
  int n = nt * 16 + (lane & 15);
  d[idx] = (_Float16)w[k * N + n];
}
__global__ __launch_bounds__(256) void k_packA(const float* __restrict__ w,
                                               _Float16* __restrict__ d,
                                               int M, int K) {
  int Kt = K >> 5;
  int idx = blockIdx.x * 256 + threadIdx.x;
  if (idx >= M * K) return;
  int i = idx & 15;
  int lane = (idx >> 4) & 31;
  int tile = idx >> 9;
  int kt = tile % Kt, mt = tile / Kt;
  int m = mt * 16 + (lane & 15);
  int k = kt * 32 + ((lane >> 4) << 4) + i;
  d[idx] = (_Float16)w[m * K + k];
}

// ================= K1: 1x1 conv (pre1) as WMMA GEMM =================
// One wave per 16-pixel tile; 12 accumulators cover all 192 output channels,
// so each x fragment is loaded once and reused by 12 WMMAs.
__global__ __launch_bounds__(256) void k_pre1(const float* __restrict__ x,
                                              const _Float16* __restrict__ wA,
                                              const float* __restrict__ bias,
                                              float* __restrict__ f) {
  int wave = threadIdx.x >> 5, lane = threadIdx.x & 31;
  int tile = blockIdx.x * 8 + wave;      // 0 .. 12543
  int b = tile / (PIX / 16);
  int pix0 = (tile % (PIX / 16)) * 16;
  const float* xb = x + (size_t)b * CCH * PIX + pix0;
  v8f acc[12];
#pragma unroll
  for (int o = 0; o < 12; ++o) acc[o] = zero8();
  for (int k0 = 0; k0 < CCH; k0 += 32) {
    v16h bf = ldb_g32(xb + (size_t)k0 * PIX, PIX, lane);
    int kt = k0 >> 5;
#pragma unroll
    for (int ot = 0; ot < 12; ++ot)
      acc[ot] = wmma_f16(ldfrag(wA + ((ot * 6 + kt) << 9), lane), bf, acc[ot]);
  }
  int col = lane & 15, rbse = (lane >> 4) * 8;
  float* fb = f + (size_t)b * CCH * PIX + pix0;
#pragma unroll
  for (int ot = 0; ot < 12; ++ot) {
#pragma unroll
    for (int r = 0; r < 8; ++r) {
      int o = ot * 16 + rbse + r;
      fb[(size_t)o * PIX + col] = acc[ot][r] + bias[o];
    }
  }
}

// ======== K2: depthwise 3x3 + bias + LayerNorm(C) + window partition ========
__global__ __launch_bounds__(256) void k_dwln(const float* __restrict__ f,
                                              const float* __restrict__ wdw,
                                              const float* __restrict__ bdw,
                                              const float* __restrict__ lng,
                                              const float* __restrict__ lnb,
                                              float* __restrict__ p) {
  int wave = threadIdx.x >> 5, lane = threadIdx.x & 31;
  size_t pixel = (size_t)blockIdx.x * 8 + wave;     // 0 .. 802815
  int b = (int)(pixel / PIX);
  int hw = (int)(pixel % PIX);
  int h = hw / 224, w = hw % 224;
  const float* fb = f + (size_t)b * CCH * PIX;
  float v[6];
  float s = 0.f, s2 = 0.f;
#pragma unroll
  for (int i = 0; i < 6; ++i) {
    int c = lane + 32 * i;
    float a = bdw[c];
    const float* fc = fb + (size_t)c * PIX;
    const float* wc = wdw + c * 9;
#pragma unroll
    for (int dy = -1; dy <= 1; ++dy) {
      int hh = h + dy;
      if (hh < 0 || hh >= 224) continue;
#pragma unroll
      for (int dx = -1; dx <= 1; ++dx) {
        int ww = w + dx;
        if (ww < 0 || ww >= 224) continue;
        a += wc[(dy + 1) * 3 + (dx + 1)] * fc[hh * 224 + ww];
      }
    }
    v[i] = a; s += a; s2 += a * a;
  }
#pragma unroll
  for (int m = 16; m >= 1; m >>= 1) {
    s  += __shfl_xor(s,  m, 32);
    s2 += __shfl_xor(s2, m, 32);
  }
  float mu  = s * (1.0f / 192.0f);
  float var = s2 * (1.0f / 192.0f) - mu * mu;
  float inv = rsqrtf(var + 1e-6f);
  int n = (b * 28 + (h >> 3)) * 28 + (w >> 3);
  int t = (h & 7) * 8 + (w & 7);
  float* pd = p + ((size_t)n * TOK + t) * CCH;
#pragma unroll
  for (int i = 0; i < 6; ++i) {
    int c = lane + 32 * i;
    pd[c] = (v[i] - mu) * inv * lng[c] + lnb[c];
  }
}

// ===== K3: fused per-window qkv + attention + GMLP + proj + recon =====
// 1 workgroup (8 waves) per window. 216 KB LDS, all operands in packed
// fragment layout -> every LDS fragment read is 2x ds_load_b128.
__global__ __launch_bounds__(256) void k_win(
    const float* __restrict__ p, const float* __restrict__ x,
    const _Float16* __restrict__ wqkv, const float* __restrict__ b_qkv,
    const _Float16* __restrict__ wproj, const float* __restrict__ b_proj,
    const _Float16* __restrict__ wg1, const float* __restrict__ b_g1,
    const _Float16* __restrict__ wg2, const float* __restrict__ b_g2,
    const _Float16* __restrict__ wrec, const float* __restrict__ b_rec,
    float* __restrict__ out, float* __restrict__ loss_part) {
  __shared__ __align__(16) unsigned char smem[216 * 1024];
  _Float16* pA   = (_Float16*)smem;                   // 64x192 A (Kt=6): p, later g
  _Float16* qsA  = (_Float16*)(smem + 24 * 1024);     // q, A-layout (Kt=6)
  _Float16* ksB  = (_Float16*)(smem + 48 * 1024);     // k^T, B-tiles per head
  _Float16* vsA  = (_Float16*)(smem + 72 * 1024);     // v, A-layout (Kt=6)
  _Float16* vsB  = (_Float16*)(smem + 96 * 1024);     // v, B-layout (Nt=12)
  _Float16* gmid = (_Float16*)(smem + 120 * 1024);    // 64x768 A (Kt=24), 96 KB
  float*    sc   = (float*)   (smem + 120 * 1024);    // 64x64 f32 scores (reuse)
  _Float16* prA  = (_Float16*)(smem + 136 * 1024);    // probs, A-layout (Kt=2)
  _Float16* aoA  = (_Float16*)(smem + 144 * 1024);    // attn out, A (Kt=6)
  _Float16* fuA  = (_Float16*)(smem + 168 * 1024);    // attn*g, A (Kt=6)
  _Float16* fsA  = (_Float16*)(smem + 192 * 1024);    // fused, A (Kt=6)
  float*    red  = (float*)   (smem + 120 * 1024);    // reduce scratch (reuse)

  int tid = threadIdx.x, wave = tid >> 5, lane = tid & 31;
  int n = blockIdx.x;
  int col = lane & 15, rbse = (lane >> 4) * 8;
  const float* pw = p + (size_t)n * TOK * CCH;

  // warm L2 for the packed weights (global_prefetch_b8)
  __builtin_prefetch(wqkv + (tid << 6), 0, 3);
  __builtin_prefetch(wg1 + (tid << 6), 0, 3);
  __builtin_prefetch(wg2 + (tid << 6), 0, 3);

  // Phase A: stage tokens directly into packed A-layout (Kt=6)
  for (int idx = tid; idx < TOK * CCH; idx += 256) {
    int i = idx & 15;
    int ln = (idx >> 4) & 31;
    int tile = idx >> 9;
    int kt = tile % 6, mt = tile / 6;
    int t = mt * 16 + (ln & 15);
    int c = kt * 32 + ((ln >> 4) << 4) + i;
    pA[idx] = (_Float16)pw[t * CCH + c];
  }
  __syncthreads();

  // Phase B: qkv = p @ w_qkv + b (64x576) -> scatter into q/k^T/v LDS
  for (int ti = wave; ti < 144; ti += 8) {
    int r0 = (ti & 3) * 16, c0 = (ti >> 2) * 16;
    v8f acc = zero8();
    for (int k0 = 0; k0 < CCH; k0 += 32)
      acc = wmma_f16(ldfrag(pA + (((r0 >> 4) * 6 + (k0 >> 5)) << 9), lane),
                     ldfrag(wqkv + (((k0 >> 5) * 36 + (c0 >> 4)) << 9), lane),
                     acc);
    float bias = b_qkv[c0 + col];
#pragma unroll
    for (int r = 0; r < 8; ++r) {
      int t = r0 + rbse + r;
      int cg = c0 + col;
      _Float16 val = (_Float16)(acc[r] + bias);
      if (cg < 192) {
        qsA[posA(6, t, cg)] = val;
      } else if (cg < 384) {
        int c = cg - 192;                         // k^T B-tiles, head-major
        ksB[(((c >> 5) * 4 + (t >> 4)) << 9) +
            (((((c >> 4) & 1) << 4) | (t & 15)) << 4) + (c & 15)] = val;
      } else {
        int c = cg - 384;
        vsA[posA(6, t, c)] = val;
        vsB[posB(12, t, c)] = val;
      }
    }
  }
  __syncthreads();

  // Phase C1: mid = GELU(v @ w_g1 + b_g1)  (64x768, Kt=24 A-layout)
  for (int ti = wave; ti < 192; ti += 8) {
    int r0 = (ti & 3) * 16, c0 = (ti >> 2) * 16;
    v8f acc = zero8();
    for (int k0 = 0; k0 < CCH; k0 += 32)
      acc = wmma_f16(ldfrag(vsA + (((r0 >> 4) * 6 + (k0 >> 5)) << 9), lane),
                     ldfrag(wg1 + (((k0 >> 5) * 48 + (c0 >> 4)) << 9), lane),
                     acc);
    float bias = b_g1[c0 + col];
#pragma unroll
    for (int r = 0; r < 8; ++r) {
      float z = acc[r] + bias;
      z = 0.5f * z * (1.0f + erff(z * 0.70710678118654752f));  // exact GELU
      gmid[posA(24, r0 + rbse + r, c0 + col)] = (_Float16)z;
    }
  }
  __syncthreads();

  // Phase C2: g = mid @ w_g2 + b_g2 (64x192) -> reuse pA region (A-layout)
  for (int ti = wave; ti < 48; ti += 8) {
    int r0 = (ti & 3) * 16, c0 = (ti >> 2) * 16;
    v8f acc = zero8();
    for (int k0 = 0; k0 < 768; k0 += 32)
      acc = wmma_f16(ldfrag(gmid + (((r0 >> 4) * 24 + (k0 >> 5)) << 9), lane),
                     ldfrag(wg2 + (((k0 >> 5) * 12 + (c0 >> 4)) << 9), lane),
                     acc);
    float bias = b_g2[c0 + col];
#pragma unroll
    for (int r = 0; r < 8; ++r)
      pA[posA(6, r0 + rbse + r, c0 + col)] = (_Float16)(acc[r] + bias);
  }
  __syncthreads();

  // Phase D: attention, head by head (hd=32 -> single-WMMA scores)
  const float scale = 0.17677669529663687f;  // 32^-0.5
  for (int h = 0; h < 6; ++h) {
#pragma unroll
    for (int u = 0; u < 2; ++u) {            // 16 score tiles, 2 per wave
      int ti = wave * 2 + u;
      int r0 = (ti & 3) * 16, c0t = (ti >> 2) * 16;
      v8f acc = zero8();
      acc = wmma_f16(ldfrag(qsA + (((r0 >> 4) * 6 + h) << 9), lane),
                     ldfrag(ksB + ((h * 4 + (c0t >> 4)) << 9), lane), acc);
#pragma unroll
      for (int r = 0; r < 8; ++r)
        sc[(r0 + rbse + r) * 64 + c0t + col] = acc[r] * scale;
    }
    __syncthreads();
    if (tid < 64) {                           // row softmax -> packed A-layout
      const float* row = sc + tid * 64;
      float mx = row[0];
      for (int j = 1; j < 64; ++j) mx = fmaxf(mx, row[j]);
      float sum = 0.f;
      for (int j = 0; j < 64; ++j) sum += __expf(row[j] - mx);
      float inv = 1.0f / sum;
      for (int j = 0; j < 64; ++j)
        prA[posA(2, tid, j)] = (_Float16)(__expf(row[j] - mx) * inv);
    }
    __syncthreads();
    {                                         // out_h = probs @ v_h, 1 tile/wave
      int r0 = (wave & 3) * 16, c0t = (wave >> 2) * 16;
      v8f acc = zero8();
#pragma unroll
      for (int k0 = 0; k0 < 64; k0 += 32)
        acc = wmma_f16(
            ldfrag(prA + (((r0 >> 4) * 2 + (k0 >> 5)) << 9), lane),
            ldfrag(vsB + (((k0 >> 5) * 12 + ((h * 32 + c0t) >> 4)) << 9), lane),
            acc);
#pragma unroll
      for (int r = 0; r < 8; ++r)
        aoA[posA(6, r0 + rbse + r, h * 32 + c0t + col)] = (_Float16)acc[r];
    }
    __syncthreads();
  }

  // Phase E: gate (same packed layout -> index-wise), proj, shortcut store
  for (int i = tid; i < TOK * CCH; i += 256) fuA[i] = aoA[i] * pA[i];
  __syncthreads();
  int b = n / 784, rem = n % 784, wh = rem / 28, wwn = rem % 28;
  const float* xw = x + (size_t)b * CCH * PIX;
  float* ow = out + (size_t)b * CCH * PIX;
  for (int ti = wave; ti < 48; ti += 8) {
    int r0 = (ti & 3) * 16, c0 = (ti >> 2) * 16;
    v8f acc = zero8();
    for (int k0 = 0; k0 < CCH; k0 += 32)
      acc = wmma_f16(ldfrag(fuA + (((r0 >> 4) * 6 + (k0 >> 5)) << 9), lane),
                     ldfrag(wproj + (((k0 >> 5) * 12 + (c0 >> 4)) << 9), lane),
                     acc);
    float bias = b_proj[c0 + col];
#pragma unroll
    for (int r = 0; r < 8; ++r) {
      int t = r0 + rbse + r;
      int c = c0 + col;
      float val = acc[r] + bias;
      fsA[posA(6, t, c)] = (_Float16)val;
      int hh = wh * 8 + (t >> 3), ww = wwn * 8 + (t & 7);
      size_t gi = (size_t)c * PIX + hh * 224 + ww;
      ow[gi] = val + xw[gi];                  // window reverse + shortcut
    }
  }
  __syncthreads();

  // Phase F: recon + per-window L1 partial (deterministic)
  float lsum = 0.f;
  for (int ti = wave; ti < 48; ti += 8) {
    int r0 = (ti & 3) * 16, c0 = (ti >> 2) * 16;
    v8f acc = zero8();
    for (int k0 = 0; k0 < CCH; k0 += 32)
      acc = wmma_f16(ldfrag(fsA + (((r0 >> 4) * 6 + (k0 >> 5)) << 9), lane),
                     ldfrag(wrec + (((k0 >> 5) * 12 + (c0 >> 4)) << 9), lane),
                     acc);
    float bias = b_rec[c0 + col];
#pragma unroll
    for (int r = 0; r < 8; ++r) {
      int t = r0 + rbse + r, c = c0 + col;
      lsum += fabsf(acc[r] + bias - pw[t * CCH + c]);
    }
  }
#pragma unroll
  for (int m = 16; m >= 1; m >>= 1) lsum += __shfl_xor(lsum, m, 32);
  if (lane == 0) red[wave] = lsum;
  __syncthreads();
  if (tid == 0) {
    float tot = 0.f;
    for (int i = 0; i < 8; ++i) tot += red[i];
    loss_part[n] = tot;
  }
}

// ===== K4: fixed-order loss reduction (replay-deterministic) =====
__global__ __launch_bounds__(256) void k_loss(const float* __restrict__ part,
                                              float* __restrict__ outv) {
  __shared__ float sm[8];
  float s = 0.f;
  for (int i = threadIdx.x; i < NWIN; i += 256) s += part[i];
#pragma unroll
  for (int m = 16; m >= 1; m >>= 1) s += __shfl_xor(s, m, 32);
  int wave = threadIdx.x >> 5, lane = threadIdx.x & 31;
  if (lane == 0) sm[wave] = s;
  __syncthreads();
  if (threadIdx.x == 0) {
    float t = 0.f;
    for (int i = 0; i < 8; ++i) t += sm[i];
    outv[0] = t * (0.1f / 38535168.0f);   // RECON_W * mean
  }
}

extern "C" void kernel_launch(void* const* d_in, const int* in_sizes, int n_in,
                              void* d_out, int out_size, void* d_ws, size_t ws_size,
                              hipStream_t stream) {
  (void)in_sizes; (void)n_in; (void)out_size; (void)ws_size;
  const float* x      = (const float*)d_in[0];
  const float* w_pre1 = (const float*)d_in[1];
  const float* b_pre1 = (const float*)d_in[2];
  const float* w_dw   = (const float*)d_in[3];
  const float* b_dw   = (const float*)d_in[4];
  const float* ln_g   = (const float*)d_in[5];
  const float* ln_b   = (const float*)d_in[6];
  const float* w_qkv  = (const float*)d_in[7];
  const float* b_qkv  = (const float*)d_in[8];
  const float* w_proj = (const float*)d_in[9];
  const float* b_proj = (const float*)d_in[10];
  const float* w_g1   = (const float*)d_in[11];
  const float* b_g1   = (const float*)d_in[12];
  const float* w_g2   = (const float*)d_in[13];
  const float* b_g2   = (const float*)d_in[14];
  const float* w_rec  = (const float*)d_in[15];
  const float* b_rec  = (const float*)d_in[16];
  float* out = (float*)d_out;
  char*  ws  = (char*)d_ws;

  const size_t NELEM = (size_t)38535168;             // 4*192*224*224
  float* f_pre     = (float*)ws;                     // 154 MB
  float* p         = (float*)(ws + NELEM * 4);       // 154 MB
  float* loss_part = (float*)(ws + NELEM * 8);       // 3136 floats
  _Float16* pre1A  = (_Float16*)(ws + NELEM * 8 + 16384);  // packed weights
  _Float16* qkvB   = pre1A + 36864;                  // 192x576
  _Float16* g1B    = qkvB + 110592;                  // 192x768
  _Float16* g2B    = g1B + 147456;                   // 768x192
  _Float16* projB  = g2B + 147456;                   // 192x192
  _Float16* recB   = projB + 36864;                  // 192x192

  // one-time weight pack (cheap: 0.5M elements total)
  k_packA<<<(192 * 192 + 255) / 256, 256, 0, stream>>>(w_pre1, pre1A, 192, 192);
  k_packB<<<(192 * 576 + 255) / 256, 256, 0, stream>>>(w_qkv, qkvB, 192, 576);
  k_packB<<<(192 * 768 + 255) / 256, 256, 0, stream>>>(w_g1, g1B, 192, 768);
  k_packB<<<(768 * 192 + 255) / 256, 256, 0, stream>>>(w_g2, g2B, 768, 192);
  k_packB<<<(192 * 192 + 255) / 256, 256, 0, stream>>>(w_proj, projB, 192, 192);
  k_packB<<<(192 * 192 + 255) / 256, 256, 0, stream>>>(w_rec, recB, 192, 192);

  k_pre1<<<1568, 256, 0, stream>>>(x, pre1A, b_pre1, f_pre);
  k_dwln<<<100352, 256, 0, stream>>>(f_pre, w_dw, b_dw, ln_g, ln_b, p);
  k_win<<<NWIN, 256, 0, stream>>>(p, x, qkvB, b_qkv, projB, b_proj,
                                  g1B, b_g1, g2B, b_g2, recB, b_rec,
                                  out, loss_part);
  k_loss<<<1, 256, 0, stream>>>(loss_part, out + NELEM);
}
